// MainModel_63556926046496
// MI455X (gfx1250) — matrix operand order repaired and checked by reading, compiled, not run on hardware
//
#include <hip/hip_runtime.h>
#include <hip/hip_bf16.h>

// ---- model constants (match reference) ----
#define DIMM   1024
#define HEADS  16
#define HEADD  64
#define FFN_H  4096
#define LE     8
#define TOPK   2
#define DIM_E  512
#define DIM_S  1024
#define SEQ    2048
#define NTOK   2048   // B*S

typedef __attribute__((ext_vector_type(16))) _Float16 v16h;
typedef __attribute__((ext_vector_type(4)))  _Float16 v4h;
typedef __attribute__((ext_vector_type(8)))  float    v8f;

__device__ __forceinline__ v8f wmma16(v16h a, v16h b, v8f c) {
  // D = A(16x32 f16) * B(32x16 f16) + C(16x16 f32)   [wave32]
  return __builtin_amdgcn_wmma_f32_16x16x32_f16(false, a, false, b, (short)0, c,
                                                false, false);
}

// =====================================================================
// GEMM: C[M,N] = A[M,K] * B (+ Cadd)
//   TRANS=false: B[k*ldb + n]   TRANS=true: B[n*ldb + k]
// Requires M%128==0, N%64==0, K%32==0 (true for every call site here).
// block tile 128x64, 8 waves, each wave 32x32 (2x2 WMMA frags), K step 32
// =====================================================================
#define BM 128
#define BN 64
#define BK 32

template <bool TRANS>
__global__ __launch_bounds__(256) void gemm_kernel(
    const float* __restrict__ A, const float* __restrict__ B,
    const float* __restrict__ Cadd, float* __restrict__ C,
    int M, int N, int K, int lda, int ldb, int ldc, int addC)
{
  __shared__ _Float16 As[BM * BK];   // [m][k]
  __shared__ _Float16 Bs[BK * BN];   // NN: [k][n]   NT: [n][k]  (same size)
  const int tid  = threadIdx.x;
  const int wave = tid >> 5;
  const int lane = tid & 31;
  const int half = lane >> 4;
  const int l16  = lane & 15;
  const int wm   = wave >> 1;   // 0..3
  const int wn   = wave & 1;    // 0..1
  const int bm   = blockIdx.y * BM;
  const int bn   = blockIdx.x * BN;

  v8f zero = {};
  v8f acc[2][2];
  for (int i = 0; i < 2; ++i) for (int j = 0; j < 2; ++j) acc[i][j] = zero;

  for (int k0 = 0; k0 < K; k0 += BK) {
    // ---- stage A tile: 128x32 f32 -> f16, float4 loads, b64 LDS stores ----
#pragma unroll
    for (int i = 0; i < 4; ++i) {
      int v = tid + i * 256;           // 0..1023 float4 slots
      int r  = v >> 3;                 // row in tile
      int kc = (v & 7) << 2;           // k offset (x4)
      float4 f = *(const float4*)(A + (long)(bm + r) * lda + k0 + kc);
      v4h h; h[0] = (_Float16)f.x; h[1] = (_Float16)f.y;
             h[2] = (_Float16)f.z; h[3] = (_Float16)f.w;
      *(v4h*)(&As[r * BK + kc]) = h;
    }
    // ---- stage B tile ----
#pragma unroll
    for (int i = 0; i < 2; ++i) {
      int v = tid + i * 256;           // 0..511 float4 slots
      if (TRANS) {
        int n  = v >> 3;               // col in tile
        int kc = (v & 7) << 2;
        float4 f = *(const float4*)(B + (long)(bn + n) * ldb + k0 + kc);
        v4h h; h[0] = (_Float16)f.x; h[1] = (_Float16)f.y;
               h[2] = (_Float16)f.z; h[3] = (_Float16)f.w;
        *(v4h*)(&Bs[n * BK + kc]) = h;
      } else {
        int kk = v >> 4;
        int nc = (v & 15) << 2;
        float4 f = *(const float4*)(B + (long)(k0 + kk) * ldb + bn + nc);
        v4h h; h[0] = (_Float16)f.x; h[1] = (_Float16)f.y;
               h[2] = (_Float16)f.z; h[3] = (_Float16)f.w;
        *(v4h*)(&Bs[kk * BN + nc]) = h;
      }
    }
    // prefetch next K tile (global_prefetch_b8 path)
    if (k0 + BK < K) {
      __builtin_prefetch(A + (long)(bm + (tid >> 1)) * lda + k0 + BK, 0, 3);
      if (TRANS) __builtin_prefetch(B + (long)(bn + (tid >> 2)) * ldb + k0 + BK, 0, 3);
      else       __builtin_prefetch(B + (long)(k0 + BK + (tid >> 3)) * ldb + bn, 0, 3);
    }
    __syncthreads();

    v16h af[2], bf[2];
#pragma unroll
    for (int sm = 0; sm < 2; ++sm) {
      int row = wm * 32 + sm * 16 + l16;
#pragma unroll
      for (int j = 0; j < 16; ++j) {
        int kk = half * 8 + (j >> 3) * 16 + (j & 7);
        af[sm][j] = As[row * BK + kk];
      }
    }
#pragma unroll
    for (int sn = 0; sn < 2; ++sn) {
      int col = wn * 32 + sn * 16 + l16;
#pragma unroll
      for (int j = 0; j < 16; ++j) {
        int kk = half * 8 + (j >> 3) * 16 + (j & 7);
        bf[sn][j] = TRANS ? Bs[col * BK + kk] : Bs[kk * BN + col];
      }
    }
#pragma unroll
    for (int sm = 0; sm < 2; ++sm)
#pragma unroll
      for (int sn = 0; sn < 2; ++sn)
        acc[sm][sn] = wmma16(af[sm], bf[sn], acc[sm][sn]);
    __syncthreads();
  }

#pragma unroll
  for (int sm = 0; sm < 2; ++sm)
#pragma unroll
    for (int sn = 0; sn < 2; ++sn)
#pragma unroll
      for (int r = 0; r < 8; ++r) {
        int row = bm + wm * 32 + sm * 16 + r + half * 8;
        int col = bn + wn * 32 + sn * 16 + l16;
        float v = acc[sm][sn][r];
        if (addC) v += Cadd[(long)row * ldc + col];
        C[(long)row * ldc + col] = v;
      }
}

// =====================================================================
// Flash attention: 1 wave per (16-row q tile, head). causal + same-doc mask.
// qkv layout [SEQ, 3*DIMM]; out [SEQ, DIMM]
// =====================================================================
__global__ __launch_bounds__(32) void attn_kernel(
    const float* __restrict__ qkv, const int* __restrict__ doc,
    float* __restrict__ out)
{
  const int lane  = threadIdx.x;
  const int half  = lane >> 4, l16 = lane & 15;
  const int qbase = blockIdx.x * 16;
  const int head  = blockIdx.y;
  const int ld    = 3 * DIMM;
  const float qs  = 0.125f;   // 1/sqrt(64)

  __shared__ _Float16 Psm[16 * 32];

  // Q A-fragments (pre-scaled); per lane-half the k-runs are contiguous
  v16h qa[2];
  {
    const float* qp = qkv + (long)(qbase + l16) * ld + head * HEADD;
#pragma unroll
    for (int c0 = 0; c0 < 2; ++c0) {
      float4 f0 = *(const float4*)(qp + c0 * 32 + half * 16);
      float4 f1 = *(const float4*)(qp + c0 * 32 + half * 16 + 4);
      float4 f2 = *(const float4*)(qp + c0 * 32 + half * 16 + 8);
      float4 f3 = *(const float4*)(qp + c0 * 32 + half * 16 + 12);
      qa[c0][0]  = (_Float16)(f0.x * qs); qa[c0][1]  = (_Float16)(f0.y * qs);
      qa[c0][2]  = (_Float16)(f0.z * qs); qa[c0][3]  = (_Float16)(f0.w * qs);
      qa[c0][4]  = (_Float16)(f1.x * qs); qa[c0][5]  = (_Float16)(f1.y * qs);
      qa[c0][6]  = (_Float16)(f1.z * qs); qa[c0][7]  = (_Float16)(f1.w * qs);
      qa[c0][8]  = (_Float16)(f2.x * qs); qa[c0][9]  = (_Float16)(f2.y * qs);
      qa[c0][10] = (_Float16)(f2.z * qs); qa[c0][11] = (_Float16)(f2.w * qs);
      qa[c0][12] = (_Float16)(f3.x * qs); qa[c0][13] = (_Float16)(f3.y * qs);
      qa[c0][14] = (_Float16)(f3.z * qs); qa[c0][15] = (_Float16)(f3.w * qs);
    }
  }

  int docq[8];
#pragma unroll
  for (int r = 0; r < 8; ++r) docq[r] = doc[qbase + r + half * 8];

  float m[8], lsum[8];
#pragma unroll
  for (int r = 0; r < 8; ++r) { m[r] = -1e30f; lsum[r] = 0.f; }
  v8f zero = {};
  v8f o[4]; for (int cg = 0; cg < 4; ++cg) o[cg] = zero;

  const int nkb = (qbase + 16 + 31) / 32;
  for (int kb = 0; kb < nkb; ++kb) {
    v8f s[2];
#pragma unroll
    for (int t = 0; t < 2; ++t) {
      int key = kb * 32 + t * 16 + l16;
      const float* kp = qkv + (long)key * ld + DIMM + head * HEADD + half * 16;
      v16h kf0, kf1;
#pragma unroll
      for (int q4 = 0; q4 < 4; ++q4) {
        float4 f = *(const float4*)(kp + q4 * 4);
        kf0[q4*4+0] = (_Float16)f.x; kf0[q4*4+1] = (_Float16)f.y;
        kf0[q4*4+2] = (_Float16)f.z; kf0[q4*4+3] = (_Float16)f.w;
        float4 g = *(const float4*)(kp + 32 + q4 * 4);
        kf1[q4*4+0] = (_Float16)g.x; kf1[q4*4+1] = (_Float16)g.y;
        kf1[q4*4+2] = (_Float16)g.z; kf1[q4*4+3] = (_Float16)g.w;
      }
      v8f sf = zero;
      sf = wmma16(qa[0], kf0, sf);
      sf = wmma16(qa[1], kf1, sf);
      int dk = doc[key];
#pragma unroll
      for (int r = 0; r < 8; ++r) {
        int qrow = qbase + r + half * 8;
        if (key > qrow || dk != docq[r]) sf[r] = -1e30f;
      }
      s[t] = sf;
    }
    // online softmax per row (rows live across 16-lane groups)
#pragma unroll
    for (int r = 0; r < 8; ++r) {
      float v = fmaxf(s[0][r], s[1][r]);
      for (int off = 8; off >= 1; off >>= 1) v = fmaxf(v, __shfl_xor(v, off, 16));
      float nm = fmaxf(m[r], v);
      float scale = (nm > -1e29f) ? __expf(m[r] - nm) : 1.0f;
      float p0 = (s[0][r] > -1e29f) ? __expf(s[0][r] - nm) : 0.f;
      float p1 = (s[1][r] > -1e29f) ? __expf(s[1][r] - nm) : 0.f;
      float rs = p0 + p1;
      for (int off = 8; off >= 1; off >>= 1) rs += __shfl_xor(rs, off, 16);
      m[r] = nm;
      lsum[r] = lsum[r] * scale + rs;
#pragma unroll
      for (int cg = 0; cg < 4; ++cg) o[cg][r] *= scale;
      int prow = r + half * 8;
      Psm[prow * 32 + l16]      = (_Float16)p0;
      Psm[prow * 32 + 16 + l16] = (_Float16)p1;
    }
    __syncthreads();
    // reload P as A-fragment, accumulate P*V
    v16h pa;
#pragma unroll
    for (int j = 0; j < 16; ++j) {
      int kk = half * 8 + (j >> 3) * 16 + (j & 7);
      pa[j] = Psm[l16 * 32 + kk];
    }
#pragma unroll
    for (int cg = 0; cg < 4; ++cg) {
      v16h vf;
#pragma unroll
      for (int j = 0; j < 16; ++j) {
        int c = half * 8 + (j >> 3) * 16 + (j & 7);
        int key = kb * 32 + c;
        vf[j] = (_Float16)qkv[(long)key * ld + 2 * DIMM + head * HEADD + cg * 16 + l16];
      }
      o[cg] = wmma16(pa, vf, o[cg]);
    }
    __syncthreads();
  }
#pragma unroll
  for (int cg = 0; cg < 4; ++cg)
#pragma unroll
    for (int r = 0; r < 8; ++r) {
      int row = qbase + r + half * 8;
      out[(long)row * DIMM + head * HEADD + cg * 16 + l16] = o[cg][r] / lsum[r];
    }
}

// ================= elementwise / small kernels =================
__global__ __launch_bounds__(256) void rmsnorm_kernel(
    const float* __restrict__ x, const float* __restrict__ g, float* __restrict__ out)
{
  int row = blockIdx.x, tid = threadIdx.x;
  const float* xr = x + (long)row * DIMM;
  float ss = 0.f;
  for (int i = tid; i < DIMM; i += 256) { float v = xr[i]; ss += v * v; }
  for (int off = 16; off >= 1; off >>= 1) ss += __shfl_xor(ss, off, 32);
  __shared__ float wsum[8];
  if ((tid & 31) == 0) wsum[tid >> 5] = ss;
  __syncthreads();
  float tot = 0.f;
  for (int w = 0; w < 8; ++w) tot += wsum[w];
  float inv = rsqrtf(tot / (float)DIMM + 1e-6f);
  for (int i = tid; i < DIMM; i += 256) out[(long)row * DIMM + i] = xr[i] * inv * g[i];
}

__global__ void rotary_kernel(float* qkv)
{
  int gid = blockIdx.x * blockDim.x + threadIdx.x;
  if (gid >= SEQ * HEADS * 32) return;
  int i = gid & 31;
  int h = (gid >> 5) & (HEADS - 1);
  int s = gid >> 9;
  float freq = (float)s * __powf(1.0f / 10000.0f, (float)i / 32.0f);
  float sn, cs;
  __sincosf(freq, &sn, &cs);
  float* q = qkv + (long)s * 3 * DIMM + h * HEADD;
  float* k = q + DIMM;
  float x1 = q[i], x2 = q[i + 32];
  q[i] = x1 * cs + x2 * sn;  q[i + 32] = -x1 * sn + x2 * cs;
  x1 = k[i]; x2 = k[i + 32];
  k[i] = x1 * cs + x2 * sn;  k[i + 32] = -x1 * sn + x2 * cs;
}

__global__ void swiglu_kernel(const float* __restrict__ h, float* __restrict__ out,
                              int H, long total)
{
  long gid = (long)blockIdx.x * blockDim.x + threadIdx.x;
  if (gid >= total) return;
  long n = gid / H; int j = (int)(gid % H);
  float a = h[n * 2 * H + j];
  float b = h[n * 2 * H + H + j];
  out[gid] = (a / (1.f + __expf(-a))) * b;
}

__global__ void mulsilu_kernel(const float* __restrict__ h1, const float* __restrict__ h2,
                               float* __restrict__ out, long total)
{
  long gid = (long)blockIdx.x * blockDim.x + threadIdx.x;
  if (gid >= total) return;
  float a = h1[gid];
  out[gid] = (a / (1.f + __expf(-a))) * h2[gid];
}

__global__ __launch_bounds__(64) void router_kernel(
    const float* __restrict__ xf, const float* __restrict__ tkeys,
    const float* __restrict__ rbias, const long long* __restrict__ idx,
    const float* __restrict__ vals, float* __restrict__ sc)
{
  int n = blockIdx.x, tid = threadIdx.x;
  int e = tid >> 3, part = tid & 7;
  float acc = 0.f;
  const float* xr = xf + (long)n * DIMM;
  for (int d = part; d < DIMM; d += 8) acc += xr[d] * tkeys[d * LE + e];
  for (int off = 4; off >= 1; off >>= 1) acc += __shfl_xor(acc, off, 8);
  __shared__ float tv[LE];
  if (part == 0) tv[e] = acc;
  __syncthreads();
  if (tid == 0) {
    long e0 = (long)idx[n * TOPK], e1 = (long)idx[n * TOPK + 1];
    float v0 = vals[n * TOPK]     + tv[e0] + rbias[e0];
    float v1 = vals[n * TOPK + 1] + tv[e1] + rbias[e1];
    float s0 = 1.f / (1.f + __expf(-v0));
    float s1 = 1.f / (1.f + __expf(-v1));
    float inv = 1.f / (s0 + s1);
    sc[n * TOPK]     = s0 * inv;   // RSF == 1.0
    sc[n * TOPK + 1] = s1 * inv;
  }
}

__global__ void moe_combine_kernel(const float* __restrict__ ye,
    const long long* __restrict__ idx, const float* __restrict__ sc,
    float* __restrict__ y, int e)
{
  long gid = (long)blockIdx.x * blockDim.x + threadIdx.x;
  if (gid >= (long)NTOK * DIMM) return;
  long n = gid / DIMM;
  float w = 0.f;
  if ((int)idx[n * TOPK]     == e) w += sc[n * TOPK];
  if ((int)idx[n * TOPK + 1] == e) w += sc[n * TOPK + 1];
  if (w != 0.f) y[gid] += w * ye[gid];
}

// ========================= host-side orchestration =========================
struct Bufs {
  float *XCa, *XCb, *XI, *XN, *ATT, *YE, *H1, *H2, *HE, *SC, *BIG, *HSW;
};

static inline void gemm_nn(hipStream_t st, const float* A, const float* B,
                           const float* Cadd, float* C, int M, int N, int K,
                           int lda, int ldb, int ldc, int addC)
{
  dim3 grid(N / BN, M / BM);
  gemm_kernel<false><<<grid, 256, 0, st>>>(A, B, Cadd, C, M, N, K, lda, ldb, ldc, addC);
}
static inline void gemm_nt(hipStream_t st, const float* A, const float* B,
                           const float* Cadd, float* C, int M, int N, int K,
                           int lda, int ldb, int ldc, int addC)
{
  dim3 grid(N / BN, M / BM);
  gemm_kernel<true><<<grid, 256, 0, st>>>(A, B, Cadd, C, M, N, K, lda, ldb, ldc, addC);
}

static void dense_layer(hipStream_t st, const float* xin, float* xout,
    const float* aw, const float* ao, const float* up, const float* down,
    const float* ag, const float* fg, const int* doc, const Bufs& b)
{
  float* QKV = b.BIG;
  rmsnorm_kernel<<<NTOK, 256, 0, st>>>(xin, ag, b.XN);
  gemm_nn(st, b.XN, aw, nullptr, QKV, NTOK, 3 * DIMM, DIMM, DIMM, 3 * DIMM, 3 * DIMM, 0);
  rotary_kernel<<<(SEQ * HEADS * 32 + 255) / 256, 256, 0, st>>>(QKV);
  attn_kernel<<<dim3(SEQ / 16, HEADS), 32, 0, st>>>(QKV, doc, b.ATT);
  gemm_nn(st, b.ATT, ao, xin, b.XI, NTOK, DIMM, DIMM, DIMM, DIMM, DIMM, 1);
  rmsnorm_kernel<<<NTOK, 256, 0, st>>>(b.XI, fg, b.XN);
  float* H = b.BIG;   // QKV no longer needed
  gemm_nn(st, b.XN, up, nullptr, H, NTOK, 2 * FFN_H, DIMM, DIMM, 2 * FFN_H, 2 * FFN_H, 0);
  long tot = (long)NTOK * FFN_H;
  swiglu_kernel<<<(int)((tot + 255) / 256), 256, 0, st>>>(H, b.HSW, FFN_H, tot);
  gemm_nn(st, b.HSW, down, b.XI, xout, NTOK, DIMM, FFN_H, FFN_H, DIMM, DIMM, 1);
}

static void moe_layer(hipStream_t st, const float* xin, float* xout,
    const float* aw, const float* ao, const float* ag, const float* fg,
    const float* experts, const float* tkeys, const float* rbias,
    const float* sup, const float* sdown,
    const long long* idx, const float* vals, const int* doc, const Bufs& b)
{
  float* QKV = b.BIG;
  rmsnorm_kernel<<<NTOK, 256, 0, st>>>(xin, ag, b.XN);
  gemm_nn(st, b.XN, aw, nullptr, QKV, NTOK, 3 * DIMM, DIMM, DIMM, 3 * DIMM, 3 * DIMM, 0);
  rotary_kernel<<<(SEQ * HEADS * 32 + 255) / 256, 256, 0, st>>>(QKV);
  attn_kernel<<<dim3(SEQ / 16, HEADS), 32, 0, st>>>(QKV, doc, b.ATT);
  gemm_nn(st, b.ATT, ao, xin, b.XI, NTOK, DIMM, DIMM, DIMM, DIMM, DIMM, 1);
  rmsnorm_kernel<<<NTOK, 256, 0, st>>>(b.XI, fg, b.XN);           // xf
  router_kernel<<<NTOK, 64, 0, st>>>(b.XN, tkeys, rbias, idx, vals, b.SC);
  // shared expert: xout = swiglu(xf @ sup) @ sdown + xi
  float* HS = b.BIG;  // QKV done
  gemm_nn(st, b.XN, sup, nullptr, HS, NTOK, 2 * DIM_S, DIMM, DIMM, 2 * DIM_S, 2 * DIM_S, 0);
  long tot_s = (long)NTOK * DIM_S;
  swiglu_kernel<<<(int)((tot_s + 255) / 256), 256, 0, st>>>(HS, b.HSW, DIM_S, tot_s);
  gemm_nn(st, b.HSW, sdown, b.XI, xout, NTOK, DIMM, DIM_S, DIM_S, DIMM, DIMM, 1);
  // routed experts: loop e, accumulate sc-weighted contributions into xout
  const long es = (long)LE * DIMM * DIM_E;
  const float* W1 = experts;            // [LE, DIMM, DIM_E]
  const float* W2 = experts + es;
  const float* W3 = experts + 2 * es;
  long tot_e = (long)NTOK * DIM_E;
  long tot_d = (long)NTOK * DIMM;
  for (int e = 0; e < LE; ++e) {
    const float* w1 = W1 + (long)e * DIMM * DIM_E;
    const float* w2 = W2 + (long)e * DIMM * DIM_E;
    const float* w3 = W3 + (long)e * DIMM * DIM_E;
    gemm_nn(st, b.XN, w1, nullptr, b.H1, NTOK, DIM_E, DIMM, DIMM, DIM_E, DIM_E, 0);
    gemm_nn(st, b.XN, w2, nullptr, b.H2, NTOK, DIM_E, DIMM, DIMM, DIM_E, DIM_E, 0);
    mulsilu_kernel<<<(int)((tot_e + 255) / 256), 256, 0, st>>>(b.H1, b.H2, b.HE, tot_e);
    // y[n,d] = sum_h HE[n,h] * W3[d,h]  -> B transposed (contiguous along k)
    gemm_nt(st, b.HE, w3, nullptr, b.YE, NTOK, DIMM, DIM_E, DIM_E, DIM_E, DIMM, 0);
    moe_combine_kernel<<<(int)((tot_d + 255) / 256), 256, 0, st>>>(b.YE, idx, b.SC, xout, e);
  }
}

extern "C" void kernel_launch(void* const* d_in, const int* in_sizes, int n_in,
                              void* d_out, int out_size, void* d_ws, size_t ws_size,
                              hipStream_t stream)
{
  (void)in_sizes; (void)n_in; (void)out_size; (void)ws_size;
  const float*     x    = (const float*)d_in[0];
  const int*       doc  = (const int*)d_in[1];
  const long long* idx  = (const long long*)d_in[2];   // int64 indices
  const float*     vals = (const float*)d_in[3];
  const float* daw = (const float*)d_in[4];
  const float* dao = (const float*)d_in[5];
  const float* dup = (const float*)d_in[6];
  const float* ddn = (const float*)d_in[7];
  const float* dag = (const float*)d_in[8];
  const float* dfg = (const float*)d_in[9];
  const float* maw = (const float*)d_in[10];
  const float* mao = (const float*)d_in[11];
  const float* mag = (const float*)d_in[12];
  const float* mfg = (const float*)d_in[13];
  const float* mex = (const float*)d_in[14];
  const float* mtk = (const float*)d_in[15];
  const float* mrb = (const float*)d_in[16];
  const float* msu = (const float*)d_in[17];
  const float* msd = (const float*)d_in[18];

  // scratch arena (~164 MB total)
  float* w = (float*)d_ws;
  Bufs b; size_t o = 0;
  const size_t TD = (size_t)NTOK * DIMM;
  b.XCa = w + o; o += TD;
  b.XCb = w + o; o += TD;
  b.XI  = w + o; o += TD;
  b.XN  = w + o; o += TD;
  b.ATT = w + o; o += TD;
  b.YE  = w + o; o += TD;
  b.H1  = w + o; o += (size_t)NTOK * DIM_E;
  b.H2  = w + o; o += (size_t)NTOK * DIM_E;
  b.HE  = w + o; o += (size_t)NTOK * DIM_E;
  b.SC  = w + o; o += (size_t)NTOK * TOPK;
  b.BIG = w + o;                              // qkv / big FFN activations
  b.HSW = b.BIG + (size_t)NTOK * 2 * FFN_H;   // after largest BIG use

  // layer 0: dense
  dense_layer(stream, x, b.XCa, daw, dao, dup, ddn, dag, dfg, doc, b);

  // layers 1..2: MoE
  const long aw_s = (long)DIMM * 3 * DIMM, ao_s = (long)DIMM * DIMM;
  const long ex_s = 3L * LE * DIMM * DIM_E, tk_s = (long)DIMM * LE;
  const long su_s = (long)DIMM * 2 * DIM_S, sd_s = (long)DIM_S * DIMM;
  const long rt_s = (long)NTOK * TOPK;
  moe_layer(stream, b.XCa, b.XCb, maw, mao, mag, mfg, mex, mtk, mrb, msu, msd,
            idx, vals, doc, b);
  moe_layer(stream, b.XCb, b.XCa, maw + aw_s, mao + ao_s, mag + DIMM, mfg + DIMM,
            mex + ex_s, mtk + tk_s, mrb + LE, msu + su_s, msd + sd_s,
            idx + rt_s, vals + rt_s, doc, b);

  // layer 3: dense -> d_out
  dense_layer(stream, b.XCa, (float*)d_out,
              daw + aw_s, dao + ao_s, dup + (long)DIMM * 2 * FFN_H,
              ddn + (long)FFN_H * DIMM, dag + DIMM, dfg + DIMM, doc, b);
}
